// LaplacePinn_36309653520940
// MI455X (gfx1250) — matrix operand order repaired and checked by reading, not compiled, not run on hardware
//
#include <hip/hip_runtime.h>
#include <hip/hip_bf16.h>

// ---------------------------------------------------------------------------
// CDNA5 / gfx1250 fused PINN Laplace-loss kernel.
// Forward-mode 2nd-order Taylor propagation (5 streams: h, hx, hy, hxx, hyy)
// through a 2->128->128->128->1 tanh MLP. Hidden-layer matmuls run on
// v_wmma_f32_16x16x32_bf16 (bf16 A/B, f32 accumulate); all nonlinear math,
// chain-rule combines and reductions are f32.
// ---------------------------------------------------------------------------

typedef __attribute__((ext_vector_type(16))) __bf16        v16bf;
typedef __attribute__((ext_vector_type(8)))  float         v8f;
typedef __attribute__((ext_vector_type(8)))  unsigned int  v8u;

#define HDIM    128
#define PADC    136            // padded LDS row stride (ushorts) to stagger banks
#define TILE_M  16             // points per workgroup
#define NBLK_F  8192           // 131072 / 16
#define NBLK_B  256            // 4096  / 16

struct Smem {
    unsigned short wT2[HDIM * PADC];        // W2 transposed [n][k], bf16
    unsigned short wT3[HDIM * PADC];        // W3 transposed [n][k], bf16
    unsigned short act[5][TILE_M * PADC];   // streams [m][k], bf16
    float w1x[HDIM], w1y[HDIM];             // W1 rows
    float b1v[HDIM], b2v[HDIM], b3v[HDIM];
    float w4v[HDIM];
    float xs[TILE_M], ys[TILE_M];
    float red[TILE_M];
};

__device__ __forceinline__ unsigned short f2bf(float f) {
    unsigned int u = __float_as_uint(f);
    u += 0x7FFFu + ((u >> 16) & 1u);        // round-to-nearest-even
    return (unsigned short)(u >> 16);
}
__device__ __forceinline__ float bf2f(unsigned short u) {
    return __uint_as_float(((unsigned int)u) << 16);
}

// A fragment: 16x32 bf16 tile of activations, LDS layout [m][k], stride PADC.
// ISA layout: lanes 0-15 / 16-31 hold K halves 0-7 / 8-15 in v[0..3], and
// K 16-23 / 24-31 in v[4..7] (pairs packed low/high per 32-bit reg).
__device__ __forceinline__ v16bf load_a_frag(const unsigned short* A, int lane, int kt) {
    const int m    = lane & 15;
    const int koff = kt * 32 + ((lane >> 4) << 3);
    const unsigned int* row = (const unsigned int*)(A + m * PADC);
    v8u u;
    const int c0 = koff >> 1;
    const int c1 = (koff + 16) >> 1;
#pragma unroll
    for (int j = 0; j < 4; ++j) u[j]     = row[c0 + j];
#pragma unroll
    for (int j = 0; j < 4; ++j) u[4 + j] = row[c1 + j];
    return __builtin_bit_cast(v16bf, u);
}

// B fragment: 32x16 bf16 tile of weights; WT is W transposed [n][k].
// Same K order as A: lanes 0-15 / 16-31 hold K 0-7 / 8-15 in v[0..3], K 16-23 / 24-31 in v[4..7].
__device__ __forceinline__ v16bf load_b_frag(const unsigned short* WT, int lane, int kt, int colbase) {
    const int ncol = colbase + (lane & 15);
    const int kb   = kt * 32 + ((lane >> 4) << 3);
    const unsigned int* row = (const unsigned int*)(WT + ncol * PADC);
    v8u u;
    const int c = kb >> 1;
#pragma unroll
    for (int j = 0; j < 4; ++j) u[j] = row[c + j];
#pragma unroll
    for (int j = 0; j < 4; ++j) u[4 + j] = row[c + 8 + j];
    return __builtin_bit_cast(v16bf, u);
}

__global__ __launch_bounds__(256)
void pinn_laplace_wmma_kernel(const float* __restrict__ x_f, const float* __restrict__ y_f,
                              const float* __restrict__ x_b, const float* __restrict__ y_b,
                              const float* __restrict__ u_b,
                              const float* __restrict__ W1, const float* __restrict__ b1,
                              const float* __restrict__ W2, const float* __restrict__ b2,
                              const float* __restrict__ W3, const float* __restrict__ b3,
                              const float* __restrict__ W4, const float* __restrict__ b4,
                              float* __restrict__ ws)
{
    extern __shared__ char smem_raw[];
    Smem& S = *reinterpret_cast<Smem*>(smem_raw);

    const int  tid      = threadIdx.x;
    const int  wave     = tid >> 5;
    const int  lane     = tid & 31;
    const int  blk      = blockIdx.x;
    const bool isPde    = (blk < NBLK_F);
    const int  tileBase = isPde ? blk * TILE_M : (blk - NBLK_F) * TILE_M;

    // ---- stage weights (f32 -> bf16, transposed to [n][k]) ----
    for (int idx = tid; idx < HDIM * HDIM; idx += 256) {
        const int k = idx >> 7, ncol = idx & (HDIM - 1);
        S.wT2[ncol * PADC + k] = f2bf(W2[idx]);
        S.wT3[ncol * PADC + k] = f2bf(W3[idx]);
    }
    if (tid < HDIM) {
        S.w1x[tid] = W1[tid];            // W1[0][n]
        S.w1y[tid] = W1[HDIM + tid];     // W1[1][n]
        S.b1v[tid] = b1[tid];
        S.b2v[tid] = b2[tid];
        S.b3v[tid] = b3[tid];
        S.w4v[tid] = W4[tid];
    }
    if (tid < TILE_M) {
        const int p = tileBase + tid;
        S.xs[tid] = isPde ? x_f[p] : x_b[p];
        S.ys[tid] = isPde ? y_f[p] : y_b[p];
    }
    __syncthreads();

    // Lane mapping shared with the WMMA 16x16 f32 C/D layout:
    //   column n = lane&15 (+ wave slice), rows m = ((lane>>4)*8 + r) in reg r.
    const int nl    = lane & 15;
    const int ncol  = wave * 16 + nl;
    const int mbase = (lane >> 4) << 3;

    // ---- layer 1 (2 -> 128): analytic value + 1st/2nd derivative streams ----
    {
        const float wxn = S.w1x[ncol], wyn = S.w1y[ncol], bn = S.b1v[ncol];
#pragma unroll
        for (int r = 0; r < 8; ++r) {
            const int   m = mbase + r;
            const float z = S.xs[m] * wxn + S.ys[m] * wyn + bn;
            const float t = tanhf(z);
            const float g = 1.f - t * t;
            const float hx = g * wxn, hy = g * wyn;
            const float hxx = -2.f * t * wxn * hx;    // g*z'' - 2tg z'^2, z''=0
            const float hyy = -2.f * t * wyn * hy;
            const int o = m * PADC + ncol;
            S.act[0][o] = f2bf(t);
            S.act[1][o] = f2bf(hx);
            S.act[2][o] = f2bf(hy);
            S.act[3][o] = f2bf(hxx);
            S.act[4][o] = f2bf(hyy);
        }
    }
    __syncthreads();

    // ---- hidden layers 2 & 3 (128 -> 128): 5 streams via WMMA ----
    for (int lyr = 0; lyr < 2; ++lyr) {
        const unsigned short* wT = lyr ? S.wT3 : S.wT2;
        const float*          bv = lyr ? S.b3v : S.b2v;

        v8f acc0 = {}, acc1 = {}, acc2 = {}, acc3 = {}, acc4 = {};
#pragma unroll
        for (int kt = 0; kt < 4; ++kt) {
            const v16bf bfr = load_b_frag(wT, lane, kt, wave * 16);
            v16bf a;
            a = load_a_frag(S.act[0], lane, kt);
            acc0 = __builtin_amdgcn_wmma_f32_16x16x32_bf16(false, a, false, bfr, (short)0, acc0, false, false);
            a = load_a_frag(S.act[1], lane, kt);
            acc1 = __builtin_amdgcn_wmma_f32_16x16x32_bf16(false, a, false, bfr, (short)0, acc1, false, false);
            a = load_a_frag(S.act[2], lane, kt);
            acc2 = __builtin_amdgcn_wmma_f32_16x16x32_bf16(false, a, false, bfr, (short)0, acc2, false, false);
            a = load_a_frag(S.act[3], lane, kt);
            acc3 = __builtin_amdgcn_wmma_f32_16x16x32_bf16(false, a, false, bfr, (short)0, acc3, false, false);
            a = load_a_frag(S.act[4], lane, kt);
            acc4 = __builtin_amdgcn_wmma_f32_16x16x32_bf16(false, a, false, bfr, (short)0, acc4, false, false);
        }

        // Bias + tanh chain rule — elementwise in (m,n), i.e. per lane/register.
        float nh[8], nhx[8], nhy[8], nhxx[8], nhyy[8];
        const float bn = bv[ncol];
#pragma unroll
        for (int r = 0; r < 8; ++r) {
            const float z = acc0[r] + bn;
            const float t = tanhf(z);
            const float g = 1.f - t * t;
            nh[r]   = t;
            nhx[r]  = g * acc1[r];
            nhy[r]  = g * acc2[r];
            nhxx[r] = g * acc3[r] - 2.f * t * acc1[r] * nhx[r];  // g z'' - 2tg z'^2
            nhyy[r] = g * acc4[r] - 2.f * t * acc2[r] * nhy[r];
        }
        __syncthreads();   // all fragment reads done before overwriting act
#pragma unroll
        for (int r = 0; r < 8; ++r) {
            const int o = (mbase + r) * PADC + ncol;
            S.act[0][o] = f2bf(nh[r]);
            S.act[1][o] = f2bf(nhx[r]);
            S.act[2][o] = f2bf(nhy[r]);
            S.act[3][o] = f2bf(nhxx[r]);
            S.act[4][o] = f2bf(nhyy[r]);
        }
        __syncthreads();
    }

    // ---- output layer (128 -> 1) + per-block pre-scaled loss partial ----
    if (tid < TILE_M) {
        const int m = tid;
        float sum = 0.f;
        if (isPde) {
            for (int k = 0; k < HDIM; ++k)
                sum += (bf2f(S.act[3][m * PADC + k]) + bf2f(S.act[4][m * PADC + k])) * S.w4v[k];
            S.red[m] = sum * sum * (1.f / 131072.f);          // res^2 / N_f
        } else {
            for (int k = 0; k < HDIM; ++k)
                sum += bf2f(S.act[0][m * PADC + k]) * S.w4v[k];
            const float u = sum + b4[0];
            const float e = u - u_b[tileBase + m];
            S.red[m] = e * e * (1.f / 4096.f);                // err^2 / N_b
        }
    }
    __syncthreads();
    if (tid == 0) {
        float s = 0.f;
        for (int i = 0; i < TILE_M; ++i) s += S.red[i];
        ws[blk] = s;
    }
}

// Deterministic fixed-order final reduction: d_out[0] = sum(ws[0..n-1]).
__global__ __launch_bounds__(256)
void pinn_reduce_kernel(const float* __restrict__ ws, float* __restrict__ out, int n)
{
    __shared__ float red[256];
    float s = 0.f;
    for (int i = threadIdx.x; i < n; i += 256) s += ws[i];
    red[threadIdx.x] = s;
    __syncthreads();
    for (int off = 128; off > 0; off >>= 1) {
        if ((int)threadIdx.x < off) red[threadIdx.x] += red[threadIdx.x + off];
        __syncthreads();
    }
    if (threadIdx.x == 0) out[0] = red[0];
}

extern "C" void kernel_launch(void* const* d_in, const int* in_sizes, int n_in,
                              void* d_out, int out_size, void* d_ws, size_t ws_size,
                              hipStream_t stream) {
    const float* x_f = (const float*)d_in[0];
    const float* y_f = (const float*)d_in[1];
    const float* x_b = (const float*)d_in[2];
    const float* y_b = (const float*)d_in[3];
    const float* u_b = (const float*)d_in[4];
    const float* W1  = (const float*)d_in[5];
    const float* b1  = (const float*)d_in[6];
    const float* W2  = (const float*)d_in[7];
    const float* b2  = (const float*)d_in[8];
    const float* W3  = (const float*)d_in[9];
    const float* b3  = (const float*)d_in[10];
    const float* W4  = (const float*)d_in[11];
    const float* b4  = (const float*)d_in[12];
    float* ws  = (float*)d_ws;
    float* out = (float*)d_out;

    const int nblk = NBLK_F + NBLK_B;       // 8448 blocks, 16 points each
    const size_t smem = sizeof(Smem);       // ~92 KB dynamic LDS (<< 320 KB/WGP)

    pinn_laplace_wmma_kernel<<<nblk, 256, smem, stream>>>(
        x_f, y_f, x_b, y_b, u_b, W1, b1, W2, b2, W3, b3, W4, b4, ws);
    pinn_reduce_kernel<<<1, 256, 0, stream>>>(ws, out, nblk);
}
